// Model_LSTM_7997229105627
// MI455X (gfx1250) — matrix operand order repaired and checked by reading, not compiled, not run on hardware
//
#include <hip/hip_runtime.h>

#define H_DIM 100
#define NPAD  112          // 7 tiles * 16 per gate
#define KPAD  128          // 4 k-steps * 32
#define NT    7
#define WAVES 8
#define ROWS_PER_WAVE 16
#define ROWS_PER_BLOCK (WAVES * ROWS_PER_WAVE)   // 128
#define BATCH 65536
#define SEQ_IN 5
#define SEQ_OUT 8

#define WB_HALVES (NT * 4 * 4 * 32 * 16)   // 57344 halves = 114688 bytes
#define H_HALVES  (16 * KPAD)              // 2048 halves = 4096 bytes per wave
#define SMEM_BYTES (WB_HALVES * 2 + WAVES * H_HALVES * 2 + WAVES * 16 * 4) // 147968

typedef __attribute__((ext_vector_type(16))) _Float16 v16h;
typedef __attribute__((ext_vector_type(8)))  float    v8f;

__device__ __forceinline__ float fast_sig(float x) {
  // sigmoid(x) = 1 / (1 + 2^(-x * log2(e)))
  float e = __builtin_amdgcn_exp2f(-1.44269504f * x);
  return __builtin_amdgcn_rcpf(1.0f + e);
}
__device__ __forceinline__ float fast_tanh(float x) {
  // tanh(x) = (1 - 2^(-2x*log2(e))) / (1 + 2^(-2x*log2(e)))
  float e = __builtin_amdgcn_exp2f(-2.88539008f * x);
  return (1.0f - e) * __builtin_amdgcn_rcpf(1.0f + e);
}

// Per-wave LDS fence: Hs/XY are private to one wave, so a full workgroup
// barrier is unnecessary. Same-wave DS ops complete in order; s_wait_dscnt 0
// makes all prior LDS writes by this wave visible to its own (cross-lane)
// reads. wave_barrier() pins compiler scheduling; "memory" blocks caching.
__device__ __forceinline__ void wave_lds_sync() {
  __builtin_amdgcn_wave_barrier();
  asm volatile("s_wait_dscnt 0" ::: "memory");
  __builtin_amdgcn_wave_barrier();
}

// ---------------------------------------------------------------------------
// Prep: repack weights into WMMA-B-fragment order (per-lane contiguous) + pads
// B fragment (32x16 f16, K x N): lane L<16 half hh -> (K = kt*32 + hh, N = L)
//                                lane L>=16       -> (K = kt*32 + 16 + hh, N = L-16)
// Fragment linear index: [t][gate][kt], value = W_hh[gate*100 + t*16 + N][K]
// ---------------------------------------------------------------------------
__global__ void lstm_prep(const float* __restrict__ W_ih,
                          const float* __restrict__ W_hh,
                          const float* __restrict__ b_ih,
                          const float* __restrict__ b_hh,
                          const float* __restrict__ W_o,
                          _Float16* __restrict__ WBp,
                          float* __restrict__ wih_p,
                          float* __restrict__ bias_p,
                          float* __restrict__ wo_p) {
  int idx = blockIdx.x * blockDim.x + threadIdx.x;
  const int nfragLane = NT * 4 * 4 * 32;   // 3584
  if (idx < nfragLane) {
    int lane = idx & 31;
    int kt   = (idx >> 5) & 3;
    int gate = (idx >> 7) & 3;
    int t    = idx >> 9;
    int ln = lane & 15, hi = lane >> 4;
    _Float16* dst = WBp + idx * 16;
    int jcol = t * 16 + ln;
    int r = gate * H_DIM + jcol;
    for (int hh = 0; hh < 16; ++hh) {
      int k = kt * 32 + hi * 16 + hh;
      float v = (jcol < H_DIM && k < H_DIM) ? W_hh[r * H_DIM + k] : 0.0f;
      dst[hh] = (_Float16)v;
    }
  }
  if (idx < 4 * NPAD) {
    int gate = idx / NPAD, col = idx % NPAD;
    float w = 0.0f, b = 0.0f;
    if (col < H_DIM) {
      int r = gate * H_DIM + col;
      w = W_ih[r];
      b = b_ih[r] + b_hh[r];
    }
    wih_p[idx]  = w;
    bias_p[idx] = b;
  }
  if (idx < NPAD) {
    wo_p[idx] = (idx < H_DIM) ? W_o[idx] : 0.0f;
  }
}

// ---------------------------------------------------------------------------
// Fused encoder+decoder. One wave = 16 batch rows, resident h (LDS f16) and
// c (registers) across all 12 timesteps. 112 WMMAs per wave per step.
// ---------------------------------------------------------------------------
__global__ void __launch_bounds__(256)
lstm_fused(const float* __restrict__ x,
           const float* __restrict__ b_o,
           const _Float16* __restrict__ WBp,
           const float* __restrict__ wih_p,
           const float* __restrict__ bias_p,
           const float* __restrict__ wo_p,
           float* __restrict__ out) {
  extern __shared__ char smem[];
  _Float16* WB = (_Float16*)smem;
  const int wave = threadIdx.x >> 5;
  _Float16* Hs = (_Float16*)(smem + WB_HALVES * 2) + wave * H_HALVES;
  float* XY = (float*)(smem + WB_HALVES * 2 + WAVES * H_HALVES * 2) + wave * 16;

  const int tid  = threadIdx.x;
  const int lane = tid & 31;
  const int ln   = lane & 15;
  const int hi   = lane >> 4;
  const int row_base = blockIdx.x * ROWS_PER_BLOCK + wave * ROWS_PER_WAVE;

  // Stage packed weight fragments into LDS via the CDNA5 async path:
  // global_load_async_to_lds_b128 bypasses VGPRs, tracked by ASYNCcnt.
  // Low 32 bits of a generic pointer into LDS are the wave-relative LDS
  // byte offset (ISA aperture mapping), which is what VDST expects.
  {
    const char* gsrc = (const char*)WBp;
    const uint32_t ldsWB = (uint32_t)(uintptr_t)WB;
    for (int i = tid; i < (WB_HALVES * 2) / 16; i += 256) {
      const void* g = gsrc + (size_t)i * 16;
      uint32_t l = ldsWB + (uint32_t)i * 16;
      asm volatile("global_load_async_to_lds_b128 %0, %1, off"
                   :: "v"(l), "v"(g) : "memory");
    }
    asm volatile("s_wait_asynccnt 0" ::: "memory");
  }
  // Zero per-wave h buffer (h0 = 0, and K/N padding stays zero forever).
  {
    uint4 z = {0u, 0u, 0u, 0u};
    uint4* hz = (uint4*)Hs;
    for (int i = lane; i < H_HALVES / 8; i += 32) hz[i] = z;
  }
  __syncthreads();   // only workgroup barrier: WB is shared read-only after this

  // Per-lane column constants: column j = t*16 + ln of each gate.
  float wihR[4][NT], biasR[4][NT], woR[NT];
#pragma unroll
  for (int g = 0; g < 4; ++g)
#pragma unroll
    for (int t = 0; t < NT; ++t) {
      wihR[g][t]  = wih_p[g * NPAD + t * 16 + ln];
      biasR[g][t] = bias_p[g * NPAD + t * 16 + ln];
    }
#pragma unroll
  for (int t = 0; t < NT; ++t) woR[t] = wo_p[t * 16 + ln];
  const float bo = b_o[0];

  // Cell state in registers, D-fragment layout: cReg[t][v] is (row v+hi*8, col t*16+ln)
  float cReg[NT][8];
#pragma unroll
  for (int t = 0; t < NT; ++t)
#pragma unroll
    for (int v = 0; v < 8; ++v) cReg[t][v] = 0.0f;

  for (int s = 0; s < 12; ++s) {
    // Input for this step: encoder reads x[:, s, 0]; decoder reuses y in XY.
    if (s < SEQ_IN && lane < 16) {
      XY[lane] = x[(size_t)(row_base + lane) * SEQ_IN + s];
    }
    wave_lds_sync();

    float xr[8];
#pragma unroll
    for (int v = 0; v < 8; ++v) xr[v] = XY[v + hi * 8];

    // Load A fragments (h, 16x32 f16 each) in the same contiguous k order as packed B:
    // lanes 0-15: K = kt*32 + {0..15};  lanes 16-31: K = kt*32 + {16..31}
    v16h a[4];
#pragma unroll
    for (int kt = 0; kt < 4; ++kt) {
      const uint4* p = (const uint4*)((const char*)Hs + ln * (KPAD * 2) + kt * 64 + hi * 32);
      union { v16h v; uint4 q[2]; } u;
      u.q[0] = p[0];   // +0 bytes  (8 halves)
      u.q[1] = p[1];   // +16 bytes (8 halves)
      a[kt] = u.v;
    }

    float yp[8];
#pragma unroll
    for (int v = 0; v < 8; ++v) yp[v] = 0.0f;

#pragma unroll
    for (int t = 0; t < NT; ++t) {
      v8f acc[4];
#pragma unroll
      for (int g = 0; g < 4; ++g) {
        v8f c = {};
#pragma unroll
        for (int kt = 0; kt < 4; ++kt) {
          const v16h b = *(const v16h*)(WB + (((t * 16 + g * 4 + kt) * 32) + lane) * 16);
          c = __builtin_amdgcn_wmma_f32_16x16x32_f16(
              /*neg_a=*/false, a[kt], /*neg_b=*/false, b,
              /*c_mod=*/(short)0, c, /*reuse_a=*/false, /*reuse_b=*/false);
        }
        acc[g] = c;
      }
      // Elementwise LSTM cell on D fragments (i, f, g, o at identical positions).
#pragma unroll
      for (int v = 0; v < 8; ++v) {
        float gi = acc[0][v] + xr[v] * wihR[0][t] + biasR[0][t];
        float gf = acc[1][v] + xr[v] * wihR[1][t] + biasR[1][t];
        float gg = acc[2][v] + xr[v] * wihR[2][t] + biasR[2][t];
        float go = acc[3][v] + xr[v] * wihR[3][t] + biasR[3][t];
        float c  = fast_sig(gf) * cReg[t][v] + fast_sig(gi) * fast_tanh(gg);
        float h  = fast_sig(go) * fast_tanh(c);
        cReg[t][v] = c;
        yp[v] += h * woR[t];
        Hs[(v + hi * 8) * KPAD + t * 16 + ln] = (_Float16)h;
      }
    }

    // Output projection y = h . W_o + b_o after steps 4..11.
    if (s >= SEQ_IN - 1) {
#pragma unroll
      for (int v = 0; v < 8; ++v) {
        float y = yp[v];
        y += __shfl_xor(y, 1, 32);
        y += __shfl_xor(y, 2, 32);
        y += __shfl_xor(y, 4, 32);
        y += __shfl_xor(y, 8, 32);
        y += bo;
        if (ln == 0) {
          int row = row_base + v + hi * 8;
          out[(size_t)row * SEQ_OUT + (s - (SEQ_IN - 1))] = y;
          XY[v + hi * 8] = y;   // feeds next decoder step
        }
      }
    }
    wave_lds_sync();   // h stores + XY visible to this wave's next-step reads
  }
}

extern "C" void kernel_launch(void* const* d_in, const int* in_sizes, int n_in,
                              void* d_out, int out_size, void* d_ws, size_t ws_size,
                              hipStream_t stream) {
  const float* x    = (const float*)d_in[0];
  const float* W_ih = (const float*)d_in[1];
  const float* W_hh = (const float*)d_in[2];
  const float* b_ih = (const float*)d_in[3];
  const float* b_hh = (const float*)d_in[4];
  const float* W_o  = (const float*)d_in[5];
  const float* b_o  = (const float*)d_in[6];
  float* out = (float*)d_out;

  char* ws = (char*)d_ws;
  _Float16* WBp = (_Float16*)ws;                      // 114688 B
  float* wih_p  = (float*)(ws + 114688);              // 1792 B
  float* bias_p = (float*)(ws + 114688 + 1792);       // 1792 B
  float* wo_p   = (float*)(ws + 114688 + 2 * 1792);   // 448 B

  lstm_prep<<<14, 256, 0, stream>>>(W_ih, W_hh, b_ih, b_hh, W_o,
                                    WBp, wih_p, bias_p, wo_p);

  (void)hipFuncSetAttribute((const void*)lstm_fused,
                            hipFuncAttributeMaxDynamicSharedMemorySize,
                            (int)SMEM_BYTES);
  lstm_fused<<<BATCH / ROWS_PER_BLOCK, 256, SMEM_BYTES, stream>>>(
      x, b_o, WBp, wih_p, bias_p, wo_p, out);
}
